// DifferentiableRobustMVOLayer_27650999452181
// MI455X (gfx1250) — matrix operand order repaired and checked by reading, not compiled, not run on hardware
//
#include <hip/hip_runtime.h>

#define N 512
#define BATCH 128
#define LAMBDA_W 1.0f
#define KAPPA_W 0.1f
#define N_ITERS 400

typedef __attribute__((ext_vector_type(16))) __bf16 v16bf;
typedef __attribute__((ext_vector_type(8)))  float  v8f;

union frag_u { uint4 q[2]; v16bf v; };

__device__ inline unsigned short f2bf(float f) {
  unsigned u = __float_as_uint(f);
  u += 0x7FFFu + ((u >> 16) & 1u);          // round-to-nearest-even
  return (unsigned short)(u >> 16);
}

// fma two packed bf16 (in one dword) against two f32 weights
__device__ inline float bfpair(unsigned u, float a, float b, float acc) {
  float lo = __uint_as_float(u << 16);
  float hi = __uint_as_float(u & 0xFFFF0000u);
  return fmaf(lo, a, fmaf(hi, b, acc));
}

// ---------------------------------------------------------------------------
// Kernel 1: Gram matrices  Sigma = U^T U,  G = A^T A  via v_wmma_f32_16x16x32_bf16
// grid = (10 upper-tri 128x128 tiles, 2 matrices, 128 batches), 256 thr / 8 waves.
// Wave (wr,wc) owns a 32x64 sub-tile -> 8 f32 accumulators, 8 WMMAs / k-step.
// LDS panels stored TRANSPOSED as packed bf16-pairs: PanelT[col][k/2] so every
// WMMA fragment is two contiguous 16B chunks -> ds_load_b128 (no u16 scatter).
// Column stride 20 dwords (80 B): 16B-aligned b128 + conflict-free bank map.
// ---------------------------------------------------------------------------
__global__ __launch_bounds__(256) void gram_kernel(
    const float* __restrict__ U, const float* __restrict__ A,
    unsigned short* __restrict__ SigmaBF, unsigned short* __restrict__ GBF) {
  // triangular tile index -> (ti, tj), ti <= tj, 4x4 tile grid
  int rem = blockIdx.x, row = 0;
  while (rem >= 4 - row) { rem -= 4 - row; row++; }
  const int i0 = row * 128;
  const int j0 = (row + rem) * 128;

  const int which = blockIdx.y;            // 0: U->Sigma, 1: A->G
  const int b     = blockIdx.z;
  const float* src = (which == 0 ? U : A) + (size_t)b * N * N;
  unsigned short* dst = (which == 0 ? SigmaBF : GBF) + (size_t)b * N * N;

  __shared__ __align__(16) unsigned int ApT[128][20];  // [col][k-pair], 32 k / col
  __shared__ __align__(16) unsigned int BpT[128][20];

  const int tid   = threadIdx.x;
  const int lane  = tid & 31;
  const int wv    = tid >> 5;              // 0..7
  const int wr    = wv >> 1;               // 0..3  (M chunk of 32)
  const int wc    = wv & 1;                // 0..1  (N chunk of 64)
  const int khalf = (lane >> 4) & 1;
  const int l15   = lane & 15;

  v8f acc[2][4] = {};

  for (int kb = 0; kb < N; kb += 32) {
    // ---- load + convert + pack panels (transposed, k-pairs in dwords) ----
    #pragma unroll
    for (int i = 0; i < 8; i++) {          // 2048 dword-pairs per panel
      int p  = tid + 256 * i;
      int r2 = p >> 7;                     // k-pair index 0..15
      int c  = p & 127;
      float a0 = src[(size_t)(kb + 2 * r2)     * N + i0 + c];
      float a1 = src[(size_t)(kb + 2 * r2 + 1) * N + i0 + c];
      ApT[c][r2] = (unsigned)f2bf(a0) | ((unsigned)f2bf(a1) << 16);
      float b0 = src[(size_t)(kb + 2 * r2)     * N + j0 + c];
      float b1 = src[(size_t)(kb + 2 * r2 + 1) * N + j0 + c];
      BpT[c][r2] = (unsigned)f2bf(b0) | ((unsigned)f2bf(b1) << 16);
    }
    if (kb + 32 < N)                        // global_prefetch_b8 for next panel
      __builtin_prefetch(&src[(size_t)(kb + 32) * N + i0 + (tid & 127)], 0, 1);
    __syncthreads();

    // ---- A fragments: lane column m, k = khalf*8+{0..7} and +16 -> 2x b128 ----
    v16bf afrag[2];
    #pragma unroll
    for (int mi = 0; mi < 2; mi++) {
      int m = wr * 32 + mi * 16 + l15;
      frag_u u;
      u.q[0] = *(const uint4*)&ApT[m][khalf * 4];
      u.q[1] = *(const uint4*)&ApT[m][8 + khalf * 4];
      afrag[mi] = u.v;
    }
    // ---- B fragments: lane column n, k = khalf*8+{0..7} and +16 -> 2x b128 ----
    #pragma unroll
    for (int nb = 0; nb < 4; nb++) {
      int n = wc * 64 + nb * 16 + l15;
      frag_u u;
      u.q[0] = *(const uint4*)&BpT[n][khalf * 4];
      u.q[1] = *(const uint4*)&BpT[n][8 + khalf * 4];
      v16bf bfrag = u.v;
      acc[0][nb] = __builtin_amdgcn_wmma_f32_16x16x32_bf16(
          false, afrag[0], false, bfrag, (short)0, acc[0][nb], false, false);
      acc[1][nb] = __builtin_amdgcn_wmma_f32_16x16x32_bf16(
          false, afrag[1], false, bfrag, (short)0, acc[1][nb], false, false);
    }
    __syncthreads();
  }

  // C/D f32 16x16 layout: col = lane%16, row = r + 8*(lane/16).
  // Mirror store: accumulation order identical for (i,j)/(j,i) -> bitwise equal.
  #pragma unroll
  for (int mi = 0; mi < 2; mi++)
    #pragma unroll
    for (int nb = 0; nb < 4; nb++)
      #pragma unroll
      for (int r = 0; r < 8; r++) {
        int rw = i0 + wr * 32 + mi * 16 + khalf * 8 + r;
        int cl = j0 + wc * 64 + nb * 16 + l15;
        unsigned short val = f2bf(acc[mi][nb][r]);
        dst[(size_t)rw * N + cl] = val;
        dst[(size_t)cl * N + rw] = val;
      }
}

// ---------------------------------------------------------------------------
// Wave-cooperative 512x512 bf16 mat-vec: lane owns 16 contiguous columns
// (two uint4 loads per row), cross-lane shfl reduction. blockDim = 512.
// ---------------------------------------------------------------------------
__device__ inline void matvec512(const unsigned short* __restrict__ M,
                                 const float* __restrict__ w_lds,
                                 float* __restrict__ y_lds) {
  const int tid = threadIdx.x;
  const int lane = tid & 31;
  const int wv = tid >> 5;                  // 16 waves
  float wr[16];
  #pragma unroll
  for (int e = 0; e < 16; e++) wr[e] = w_lds[lane * 16 + e];
  for (int r = wv; r < N; r += 16) {
    const uint4* row = (const uint4*)(M + (size_t)r * N);
    uint4 p0 = row[lane * 2 + 0];
    uint4 p1 = row[lane * 2 + 1];
    float acc = 0.f;
    acc = bfpair(p0.x, wr[0],  wr[1],  acc);
    acc = bfpair(p0.y, wr[2],  wr[3],  acc);
    acc = bfpair(p0.z, wr[4],  wr[5],  acc);
    acc = bfpair(p0.w, wr[6],  wr[7],  acc);
    acc = bfpair(p1.x, wr[8],  wr[9],  acc);
    acc = bfpair(p1.y, wr[10], wr[11], acc);
    acc = bfpair(p1.z, wr[12], wr[13], acc);
    acc = bfpair(p1.w, wr[14], wr[15], acc);
    #pragma unroll
    for (int off = 16; off; off >>= 1) acc += __shfl_xor(acc, off, 32);
    if (lane == 0) y_lds[r] = acc;
  }
}

__device__ inline float block_sum(float v, float* part, float* scal) {
  const int lane = threadIdx.x & 31, wv = threadIdx.x >> 5;
  #pragma unroll
  for (int off = 16; off; off >>= 1) v += __shfl_xor(v, off, 32);
  if (lane == 0) part[wv] = v;
  __syncthreads();
  if (threadIdx.x == 0) {
    float s = 0.f;
    #pragma unroll
    for (int i = 0; i < 16; i++) s += part[i];
    *scal = s;
  }
  __syncthreads();
  return *scal;
}

// ---------------------------------------------------------------------------
// Kernel 2: power iteration on Sigma and G -> per-batch PGD step size
// ---------------------------------------------------------------------------
__global__ __launch_bounds__(512) void spectral_kernel(
    const unsigned short* __restrict__ SigmaBF,
    const unsigned short* __restrict__ GBF, float* __restrict__ stepArr) {
  const int b = blockIdx.x, tid = threadIdx.x;
  __shared__ float v[N], y[N], part[16], scal;
  const unsigned short* mats[2] = {SigmaBF + (size_t)b * N * N,
                                   GBF + (size_t)b * N * N};
  float lam[2];
  for (int m = 0; m < 2; m++) {
    v[tid] = 1.0f / sqrtf((float)N);
    __syncthreads();
    for (int it = 0; it < 30; it++) {
      matvec512(mats[m], v, y);
      __syncthreads();
      float s = block_sum(y[tid] * y[tid], part, &scal);
      float nrm = sqrtf(s);
      v[tid] = y[tid] / (nrm + 1e-12f);
      __syncthreads();
    }
    matvec512(mats[m], v, y);
    __syncthreads();
    lam[m] = block_sum(v[tid] * y[tid], part, &scal);
  }
  if (tid == 0)
    stepArr[b] = 1.0f / (LAMBDA_W * lam[0] + KAPPA_W * sqrtf(fmaxf(lam[1], 0.f)) + 1e-6f);
}

// ---------------------------------------------------------------------------
// Kernel 3: 400 projected-gradient iterations, L2-resident bf16 Grams.
// Simplex projection via barrier-free bisection inside wave 0.
// ---------------------------------------------------------------------------
__global__ __launch_bounds__(512) void pgd_kernel(
    const float* __restrict__ mu, const unsigned short* __restrict__ SigmaBF,
    const unsigned short* __restrict__ GBF, const float* __restrict__ stepArr,
    float* __restrict__ out) {
  const int b = blockIdx.x, tid = threadIdx.x;
  const int lane = tid & 31, wv = tid >> 5;
  __shared__ float w[N], Gw[N], Sw[N], vv[N], part[16], partm[16], scal[2];
  const unsigned short* G = GBF + (size_t)b * N * N;
  const unsigned short* S = SigmaBF + (size_t)b * N * N;
  const float mut = mu[(size_t)b * N + tid];
  const float step = stepArr[b];

  w[tid] = 1.0f / (float)N;
  __syncthreads();

  for (int it = 0; it < N_ITERS; it++) {
    matvec512(G, w, Gw);
    matvec512(S, w, Sw);
    __syncthreads();

    float wt = w[tid], gwt = Gw[tid], swt = Sw[tid];
    float s = block_sum(wt * gwt, part, &scal[0]);
    float nrm = sqrtf(s + 1e-8f);
    float g = -mut + KAPPA_W * gwt / nrm + LAMBDA_W * swt;
    float vt = wt - step * g;
    vv[tid] = vt;

    // sum & max of v (per-wave, then finished inside wave 0)
    float sv = vt, mv = vt;
    #pragma unroll
    for (int off = 16; off; off >>= 1) {
      sv += __shfl_xor(sv, off, 32);
      mv = fmaxf(mv, __shfl_xor(mv, off, 32));
    }
    if (lane == 0) { part[wv] = sv; partm[wv] = mv; }
    __syncthreads();

    if (wv == 0) {                          // wave 0: barrier-free bisection
      float reg[16];
      #pragma unroll
      for (int e = 0; e < 16; e++) reg[e] = vv[lane * 16 + e];
      float sa = (lane < 16) ? part[lane] : 0.f;
      float ma = (lane < 16) ? partm[lane] : -3.0e38f;
      #pragma unroll
      for (int off = 16; off; off >>= 1) {
        sa += __shfl_xor(sa, off, 32);
        ma = fmaxf(ma, __shfl_xor(ma, off, 32));
      }
      float lo = (sa - 1.0f) * (1.0f / (float)N);   // theta* >= (sum-1)/n
      float hi = ma;                                 // theta* <= max(v)
      for (int bi = 0; bi < 26; bi++) {
        float mid = 0.5f * (lo + hi);
        float f = 0.f;
        #pragma unroll
        for (int e = 0; e < 16; e++) f += fmaxf(reg[e] - mid, 0.f);
        #pragma unroll
        for (int off = 16; off; off >>= 1) f += __shfl_xor(f, off, 32);
        if (f >= 1.0f) lo = mid; else hi = mid;
      }
      if (lane == 0) scal[1] = 0.5f * (lo + hi);
    }
    __syncthreads();
    w[tid] = fmaxf(vv[tid] - scal[1], 0.f);
    __syncthreads();
  }

  float s = block_sum(w[tid], part, &scal[0]);
  out[(size_t)b * N + tid] = w[tid] / (s + 1e-12f);
}

// ---------------------------------------------------------------------------
extern "C" void kernel_launch(void* const* d_in, const int* in_sizes, int n_in,
                              void* d_out, int out_size, void* d_ws, size_t ws_size,
                              hipStream_t stream) {
  const float* mu = (const float*)d_in[0];
  const float* U  = (const float*)d_in[1];
  const float* A  = (const float*)d_in[2];
  float* out = (float*)d_out;

  const size_t matElems = (size_t)BATCH * N * N;      // 33.5M
  unsigned short* SigmaBF = (unsigned short*)d_ws;                 // 64 MB
  unsigned short* GBF     = SigmaBF + matElems;                    // 64 MB
  float* stepArr          = (float*)(GBF + matElems);              // 512 B

  gram_kernel<<<dim3(10, 2, BATCH), 256, 0, stream>>>(U, A, SigmaBF, GBF);
  spectral_kernel<<<BATCH, N, 0, stream>>>(SigmaBF, GBF, stepArr);
  pgd_kernel<<<BATCH, N, 0, stream>>>(mu, SigmaBF, GBF, stepArr, out);
}
